// Attention_35450660061694
// MI455X (gfx1250) — matrix operand order repaired and checked by reading, not compiled, not run on hardware
//
#include <hip/hip_runtime.h>
#include <hip/hip_bf16.h>
#include <stdint.h>

typedef _Float16 half_t;
typedef __attribute__((ext_vector_type(4)))  _Float16 v4h;
typedef __attribute__((ext_vector_type(8)))  _Float16 v8h;
typedef __attribute__((ext_vector_type(16))) _Float16 v16h;
typedef __attribute__((ext_vector_type(8)))  float    v8f;

#define DEV static __device__ __forceinline__

constexpr int B_    = 32;
constexpr int N_    = 576;
constexpr int DIM_  = 768;
constexpr int H_    = 12;
constexpr int DH_   = 64;
constexpr int BN_   = B_ * N_;        // 18432
constexpr int QKVC_ = 3 * DIM_;       // 2304
constexpr int NN_   = N_ * N_;        // 331776
constexpr float SCALE_ = 0.125f;      // 64^-0.5

// ---- workspace layout (bytes), all offsets 256B-aligned ----
constexpr size_t XH_OFF = 0;
constexpr size_t XH_SZ  = (size_t)BN_ * DIM_ * 2;          // 28,311,552
constexpr size_t WQ_OFF = XH_OFF + XH_SZ;
constexpr size_t WQ_SZ  = (size_t)QKVC_ * DIM_ * 2;        //  3,538,944
constexpr size_t WO_OFF = WQ_OFF + WQ_SZ;
constexpr size_t WO_SZ  = (size_t)DIM_ * DIM_ * 2;         //  1,179,648
constexpr size_t Q_OFF  = WO_OFF + WO_SZ;
constexpr size_t T_SZ   = (size_t)B_ * H_ * N_ * DH_ * 2;  // 28,311,552
constexpr size_t K_OFF  = Q_OFF + T_SZ;
constexpr size_t V_OFF  = K_OFF + T_SZ;
constexpr size_t O_OFF  = V_OFF + T_SZ;
constexpr size_t RB_OFF = O_OFF + XH_SZ;                   // relbT f16 [H,N(j),N(i)]

// ---- helpers ----
DEV v16h ld16(const half_t* p0, const half_t* p1) {
  union { v16h v; v8h h[2]; } u;
  u.h[0] = *reinterpret_cast<const v8h*>(p0);
  u.h[1] = *reinterpret_cast<const v8h*>(p1);
  return u.v;
}

DEV v8f wmma16(v16h a, v16h b, v8f c) {
  // D = A(16x32 f16) * B(32x16 f16) + C(16x16 f32)
  return __builtin_amdgcn_wmma_f32_16x16x32_f16(false, a, false, b,
                                                (short)0, c, false, false);
}

DEV v8f vzero() { v8f z = {0.f,0.f,0.f,0.f,0.f,0.f,0.f,0.f}; return z; }

DEV float rmax16(float x) {
  #pragma unroll
  for (int o = 8; o > 0; o >>= 1) x = fmaxf(x, __shfl_xor(x, o, 16));
  return x;
}
DEV float rsum16(float x) {
  #pragma unroll
  for (int o = 8; o > 0; o >>= 1) x += __shfl_xor(x, o, 16);
  return x;
}

// =====================  prep kernels  =====================

__global__ void k_cvt_x(const float* __restrict__ x, half_t* __restrict__ xh) {
  int i = blockIdx.x * blockDim.x + threadIdx.x;       // over BN*DIM/4
  float4 v = reinterpret_cast<const float4*>(x)[i];
  v4h o;
  o[0] = (half_t)v.x; o[1] = (half_t)v.y; o[2] = (half_t)v.z; o[3] = (half_t)v.w;
  reinterpret_cast<v4h*>(xh)[i] = o;
}

// src [rows=K, cols=N] f32  ->  dst [N, K] f16  (transpose + convert)
__global__ void k_tcvt(const float* __restrict__ src, half_t* __restrict__ dst,
                       int rows, int cols) {
  int i = blockIdx.x * blockDim.x + threadIdx.x;       // over rows*cols
  int k = i / cols, n = i - k * cols;
  dst[(size_t)n * rows + k] = (half_t)src[i];
}

// relbT[h, j, i] = bias_table[rel_index[i,j], h]   (f16, [H, N, N] j-major)
// Transposed so the attention kernel can read 8 consecutive i per lane (v8h).
__global__ void k_relbias(const float* __restrict__ bt, const int* __restrict__ ri,
                          half_t* __restrict__ rb) {
  int ij = blockIdx.x * blockDim.x + threadIdx.x;      // over N*N, ij = i*N + j
  int i = ij / N_, j = ij - i * N_;
  int idx = ri[ij];
  const float* p = bt + (size_t)idx * H_;
  #pragma unroll
  for (int h = 0; h < H_; ++h)
    rb[(size_t)h * NN_ + (size_t)j * N_ + i] = (half_t)p[h];
}

// =====================  GEMM 1: QKV projection  =====================
// Xh [BN,768] f16 (A, row-major) x WqT [2304,768] f16 (B, K-major).
// Wave computes a 32x64 strip (8 WMMA / 12 b128-loads per K-step); epilogue
// scatters into Q,K [BH,N,DH] (Q pre-scaled by SCALE) and Vt [BH,DH,N].
__global__ void k_gemm_qkv(const half_t* __restrict__ Xh, const half_t* __restrict__ Wt,
                           half_t* __restrict__ Q, half_t* __restrict__ K,
                           half_t* __restrict__ V) {
  const int lane = threadIdx.x & 31, wid = threadIdx.x >> 5;
  const int l16 = lane & 15, sel = lane >> 4;
  const int w = blockIdx.x * 8 + wid;      // 20736 waves
  const int mtile = w / 36;                // 0..575 (32-row tiles)
  const int nt64  = w - mtile * 36;        // 0..35  (64-col tiles)

  const half_t* arow0 = Xh + (size_t)(mtile * 32 + l16) * DIM_;
  const half_t* arow1 = arow0 + (size_t)16 * DIM_;
  const half_t* brow[4];
  #pragma unroll
  for (int nt = 0; nt < 4; ++nt)
    brow[nt] = Wt + (size_t)(nt64 * 64 + nt * 16 + l16) * DIM_;

  v8f acc[2][4];
  #pragma unroll
  for (int m = 0; m < 2; ++m)
    #pragma unroll
    for (int nt = 0; nt < 4; ++nt) acc[m][nt] = vzero();

  for (int kb = 0; kb < DIM_; kb += 32) {
    v16h a0 = ld16(arow0 + kb + 8 * sel, arow0 + kb + 16 + 8 * sel);
    v16h a1 = ld16(arow1 + kb + 8 * sel, arow1 + kb + 16 + 8 * sel);
    #pragma unroll
    for (int nt = 0; nt < 4; ++nt) {
      v16h b = ld16(brow[nt] + kb + 8 * sel, brow[nt] + kb + 16 + 8 * sel);
      acc[0][nt] = wmma16(a0, b, acc[0][nt]);
      acc[1][nt] = wmma16(a1, b, acc[1][nt]);
    }
  }

  const int r0   = mtile * 32;
  const int b    = r0 / N_;                // 576 % 32 == 0: tile is in one batch
  const int tok0 = r0 - b * N_;
  #pragma unroll
  for (int m = 0; m < 2; ++m) {
    const int tokm = tok0 + m * 16;
    #pragma unroll
    for (int nt = 0; nt < 4; ++nt) {
      const int c     = nt64 * 64 + nt * 16 + l16;
      const int which = c / DIM_;          // wave-uniform (64-aligned tiles)
      const int rem   = c - which * DIM_;
      const int h     = rem >> 6;          // wave-uniform
      const int d     = rem & 63;
      if (which == 0) {
        half_t* qp = Q + ((size_t)(b * H_ + h) * N_ + tokm + 8 * sel) * DH_ + d;
        #pragma unroll
        for (int v = 0; v < 8; ++v) qp[(size_t)v * DH_] = (half_t)(acc[m][nt][v] * SCALE_);
      } else if (which == 1) {
        half_t* kp = K + ((size_t)(b * H_ + h) * N_ + tokm + 8 * sel) * DH_ + d;
        #pragma unroll
        for (int v = 0; v < 8; ++v) kp[(size_t)v * DH_] = (half_t)acc[m][nt][v];
      } else {
        v8h pk;
        #pragma unroll
        for (int v = 0; v < 8; ++v) pk[v] = (half_t)acc[m][nt][v];
        *reinterpret_cast<v8h*>(V + ((size_t)(b * H_ + h) * DH_ + d) * N_ +
                                tokm + 8 * sel) = pk;
      }
    }
  }
}

// =====================  Flash attention core  =====================
// One wave = one (b,h,16-query-row tile). Streams keys in chunks of 32 with
// online softmax; P bounced through LDS (C-layout -> A-layout re-tiling).
__global__ void k_attn(const half_t* __restrict__ Q, const half_t* __restrict__ K,
                       const half_t* __restrict__ Vt, const half_t* __restrict__ RB,
                       half_t* __restrict__ O) {
  __shared__ alignas(16) half_t plds[8][16][32];   // 8 KB: 1 KB per wave
  const int lane = threadIdx.x & 31, wid = threadIdx.x >> 5;
  const int l16 = lane & 15, sel = lane >> 4;
  const int w  = blockIdx.x * 8 + wid;     // 13824 waves
  const int bh = w / 36;
  const int qt = w - bh * 36;
  const int h  = bh % H_;
  const int b  = bh / H_;
  const int i0 = qt * 16;

  const half_t* qrow = Q + ((size_t)bh * N_ + i0 + l16) * DH_;
  const v16h aq0 = ld16(qrow + 8 * sel,      qrow + 16 + 8 * sel);   // dhead 0..31
  const v16h aq1 = ld16(qrow + 32 + 8 * sel, qrow + 48 + 8 * sel);   // dhead 32..63

  v8f oacc[4];
  #pragma unroll
  for (int nt = 0; nt < 4; ++nt) oacc[nt] = vzero();
  float m[8], l[8];
  #pragma unroll
  for (int v = 0; v < 8; ++v) { m[v] = -__builtin_inff(); l[v] = 0.f; }

  const half_t* rbT = RB + (size_t)h * NN_;            // [j, i] layout

  for (int kb = 0; kb < N_; kb += 32) {
    // prefetch next key chunk (K rows + V columns); harmless overshoot into ws
    __builtin_prefetch(K + ((size_t)bh * N_ + kb + 32 + l16) * DH_, 0, 3);
    __builtin_prefetch(Vt + ((size_t)bh * DH_ + l16) * N_ + kb + 32, 0, 3);

    // ---- S = (Q*SCALE) K^T + bias, two 16-key tiles ----
    v8f s[2];
    #pragma unroll
    for (int t = 0; t < 2; ++t) {
      const half_t* krow = K + ((size_t)bh * N_ + kb + t * 16 + l16) * DH_;
      v16h b0 = ld16(krow + 8 * sel,      krow + 16 + 8 * sel);
      v16h b1 = ld16(krow + 32 + 8 * sel, krow + 48 + 8 * sel);
      v8f ss = wmma16(aq0, b0, vzero());
      ss = wmma16(aq1, b1, ss);
      // bias: lane's key column j = kb+t*16+l16, 8 consecutive i per lane
      v8h bb = *reinterpret_cast<const v8h*>(
          rbT + (size_t)(kb + t * 16 + l16) * N_ + i0 + 8 * sel);
      #pragma unroll
      for (int v = 0; v < 8; ++v) ss[v] += (float)bb[v];
      s[t] = ss;
    }
    // ---- online softmax over this 32-key chunk ----
    #pragma unroll
    for (int v = 0; v < 8; ++v) {
      float rm    = rmax16(fmaxf(s[0][v], s[1][v]));
      float mn    = fmaxf(m[v], rm);
      float alpha = __expf(m[v] - mn);
      float p0    = __expf(s[0][v] - mn);
      float p1    = __expf(s[1][v] - mn);
      l[v] = l[v] * alpha + rsum16(p0 + p1);
      m[v] = mn;
      #pragma unroll
      for (int nt = 0; nt < 4; ++nt) oacc[nt][v] *= alpha;
      plds[wid][v + 8 * sel][l16]      = (half_t)p0;   // C-layout -> LDS tile
      plds[wid][v + 8 * sel][16 + l16] = (half_t)p1;
    }
    // ---- O += P(16x32) V(32x64), P re-read in A-fragment layout ----
    v16h ap = ld16(&plds[wid][l16][16 * sel], &plds[wid][l16][16 * sel + 8]);
    #pragma unroll
    for (int nt = 0; nt < 4; ++nt) {
      const half_t* vrow = Vt + ((size_t)bh * DH_ + nt * 16 + l16) * N_ +
                           kb + 16 * sel;
      v16h bv = ld16(vrow, vrow + 8);
      oacc[nt] = wmma16(ap, bv, oacc[nt]);
    }
  }

  // ---- normalize & write O f16 [BN, 768] row-major ----
  #pragma unroll
  for (int v = 0; v < 8; ++v) {
    float inv = 1.0f / l[v];
    #pragma unroll
    for (int nt = 0; nt < 4; ++nt)
      O[((size_t)b * N_ + i0 + v + 8 * sel) * DIM_ + h * DH_ + nt * 16 + l16] =
          (half_t)(oacc[nt][v] * inv);
  }
}

// =====================  GEMM 2: output projection + bias  =====================
__global__ void k_gemm_out(const half_t* __restrict__ Oh, const half_t* __restrict__ Wt,
                           const float* __restrict__ bias, float* __restrict__ out) {
  const int lane = threadIdx.x & 31, wid = threadIdx.x >> 5;
  const int l16 = lane & 15, sel = lane >> 4;
  const int w = blockIdx.x * 8 + wid;      // 6912 waves
  const int mtile = w / 12;                // 0..575 (32-row tiles)
  const int nt64  = w - mtile * 12;

  const half_t* arow0 = Oh + (size_t)(mtile * 32 + l16) * DIM_;
  const half_t* arow1 = arow0 + (size_t)16 * DIM_;
  const half_t* brow[4];
  #pragma unroll
  for (int nt = 0; nt < 4; ++nt)
    brow[nt] = Wt + (size_t)(nt64 * 64 + nt * 16 + l16) * DIM_;

  v8f acc[2][4];
  #pragma unroll
  for (int m = 0; m < 2; ++m)
    #pragma unroll
    for (int nt = 0; nt < 4; ++nt) acc[m][nt] = vzero();

  for (int kb = 0; kb < DIM_; kb += 32) {
    v16h a0 = ld16(arow0 + kb + 8 * sel, arow0 + kb + 16 + 8 * sel);
    v16h a1 = ld16(arow1 + kb + 8 * sel, arow1 + kb + 16 + 8 * sel);
    #pragma unroll
    for (int nt = 0; nt < 4; ++nt) {
      v16h b = ld16(brow[nt] + kb + 8 * sel, brow[nt] + kb + 16 + 8 * sel);
      acc[0][nt] = wmma16(a0, b, acc[0][nt]);
      acc[1][nt] = wmma16(a1, b, acc[1][nt]);
    }
  }

  #pragma unroll
  for (int m = 0; m < 2; ++m) {
    #pragma unroll
    for (int nt = 0; nt < 4; ++nt) {
      const int c  = nt64 * 64 + nt * 16 + l16;
      const float bc = bias[c];
      #pragma unroll
      for (int v = 0; v < 8; ++v)
        out[(size_t)(mtile * 32 + m * 16 + v + 8 * sel) * DIM_ + c] =
            acc[m][nt][v] + bc;
    }
  }
}

// =====================  host launcher  =====================
extern "C" void kernel_launch(void* const* d_in, const int* in_sizes, int n_in,
                              void* d_out, int out_size, void* d_ws, size_t ws_size,
                              hipStream_t stream) {
  (void)in_sizes; (void)n_in; (void)out_size; (void)ws_size;
  const float* x          = (const float*)d_in[0];
  const float* w_qkv      = (const float*)d_in[1];
  const float* w_out      = (const float*)d_in[2];
  const float* b_out      = (const float*)d_in[3];
  const float* bias_table = (const float*)d_in[4];
  const int*   rel_index  = (const int*)d_in[5];
  float* out = (float*)d_out;

  char* ws = (char*)d_ws;
  half_t* Xh  = (half_t*)(ws + XH_OFF);
  half_t* WqT = (half_t*)(ws + WQ_OFF);
  half_t* WoT = (half_t*)(ws + WO_OFF);
  half_t* Qh  = (half_t*)(ws + Q_OFF);
  half_t* Kh  = (half_t*)(ws + K_OFF);
  half_t* Vh  = (half_t*)(ws + V_OFF);
  half_t* Oh  = (half_t*)(ws + O_OFF);
  half_t* RBh = (half_t*)(ws + RB_OFF);

  // prep: fp32 -> f16 conversions / transposes / bias gather expansion
  k_cvt_x  <<<(BN_*DIM_)/1024, 256, 0, stream>>>(x, Xh);
  k_tcvt   <<<(DIM_*QKVC_)/256, 256, 0, stream>>>(w_qkv, WqT, DIM_, QKVC_);
  k_tcvt   <<<(DIM_*DIM_)/256,  256, 0, stream>>>(w_out, WoT, DIM_, DIM_);
  k_relbias<<<NN_/256,          256, 0, stream>>>(bias_table, rel_index, RBh);

  // QKV projection (576 x 36 32x64 wave-tiles, 8 waves/block)
  k_gemm_qkv<<<(576*36)/8, 256, 0, stream>>>(Xh, WqT, Qh, Kh, Vh);

  // flash attention (384 bh x 36 query tiles)
  k_attn<<<(384*36)/8, 256, 0, stream>>>(Qh, Kh, Vh, RBh, Oh);

  // output projection + bias -> fp32 d_out
  k_gemm_out<<<(576*12)/8, 256, 0, stream>>>(Oh, WoT, b_out, out);
}
